// Attention_57492432224717
// MI455X (gfx1250) — 16-bit and 8-bit matrix operand order repaired and checked by reading, compiled, not run on hardware
//
#include <hip/hip_runtime.h>

typedef __attribute__((ext_vector_type(16))) _Float16 v16h;
typedef __attribute__((ext_vector_type(8)))  float    v8f;
typedef __attribute__((ext_vector_type(8)))  int      v8i;

constexpr int CB  = 2;
constexpr int CS  = 2048;
constexpr int CNH = 16;
constexpr int CHD = 64;
constexpr int CD  = 1024;
constexpr int CBS = CB * CS;

#if defined(__HIP_DEVICE_COMPILE__) && __has_builtin(__builtin_amdgcn_global_load_async_to_lds_b128)
#define USE_ASYNC_LDS 1
typedef int v4i_gcc __attribute__((vector_size(16)));
typedef __attribute__((address_space(1))) v4i_gcc as1_v4i;
typedef __attribute__((address_space(3))) v4i_gcc as3_v4i;
#else
#define USE_ASYNC_LDS 0
#endif

#if USE_ASYNC_LDS
__device__ __forceinline__ void wait_async_le2() {
#if __has_builtin(__builtin_amdgcn_s_wait_asynccnt)
  __builtin_amdgcn_s_wait_asynccnt(2);
#else
  asm volatile("s_wait_asynccnt 0x2" ::: "memory");
#endif
}
__device__ __forceinline__ void wait_async_le0() {
#if __has_builtin(__builtin_amdgcn_s_wait_asynccnt)
  __builtin_amdgcn_s_wait_asynccnt(0);
#else
  asm volatile("s_wait_asynccnt 0x0" ::: "memory");
#endif
}
#endif

template <int CTRL>
__device__ __forceinline__ float dpp_mov_f(float x) {
  return __int_as_float(
      __builtin_amdgcn_update_dpp(0, __float_as_int(x), CTRL, 0xf, 0xf, true));
}
__device__ __forceinline__ float row16_max(float x) {
  x = fmaxf(x, dpp_mov_f<0x121>(x));
  x = fmaxf(x, dpp_mov_f<0x122>(x));
  x = fmaxf(x, dpp_mov_f<0x124>(x));
  x = fmaxf(x, dpp_mov_f<0x128>(x));
  return x;
}
__device__ __forceinline__ float row16_sum(float x) {
  x += dpp_mov_f<0x121>(x);
  x += dpp_mov_f<0x122>(x);
  x += dpp_mov_f<0x124>(x);
  x += dpp_mov_f<0x128>(x);
  return x;
}

__global__ void k_pack_w(const float* __restrict__ w0, const float* __restrict__ w1,
                         const float* __restrict__ w2, const float* __restrict__ w3,
                         signed char* __restrict__ out) {
  int i = blockIdx.x * blockDim.x + threadIdx.x;
  const int N = CD * CD;
  if (i >= N) return;
  out[i]         = (signed char)__float2int_rn(w0[i]);
  out[N + i]     = (signed char)__float2int_rn(w1[i]);
  out[2 * N + i] = (signed char)__float2int_rn(w2[i]);
  out[3 * N + i] = (signed char)__float2int_rn(w3[i]);
}

__global__ void k_lengths(const int* __restrict__ mask, int* __restrict__ len) {
  __shared__ int sred[256];
  int b = blockIdx.x;
  const int* row = mask + (size_t)b * CS;
  int mn = 0x7fffffff;
  for (int i = threadIdx.x; i < CS; i += blockDim.x) mn = min(mn, row[i]);
  sred[threadIdx.x] = mn;
  __syncthreads();
  for (int st = 128; st >= 1; st >>= 1) {
    if ((int)threadIdx.x < st) sred[threadIdx.x] = min(sred[threadIdx.x], sred[threadIdx.x + st]);
    __syncthreads();
  }
  int minv = sred[0];
  __syncthreads();
  int idx = CS;
  for (int i = threadIdx.x; i < CS; i += blockDim.x) {
    if (row[i] == minv) { idx = i; break; }
  }
  sred[threadIdx.x] = idx;
  __syncthreads();
  for (int st = 128; st >= 1; st >>= 1) {
    if ((int)threadIdx.x < st) sred[threadIdx.x] = min(sred[threadIdx.x], sred[threadIdx.x + st]);
    __syncthreads();
  }
  if (threadIdx.x == 0) len[b] = sred[0];
}

__global__ void k_quant(const float* __restrict__ x, signed char* __restrict__ xq,
                        float* __restrict__ sx) {
  __shared__ float wred[8];
  __shared__ float sinv;
  int t = blockIdx.x;
  const float* row = x + (size_t)t * CD;
  float amax = 0.f;
  for (int c = threadIdx.x; c < CD; c += blockDim.x) amax = fmaxf(amax, fabsf(row[c]));
  for (int off = 16; off >= 1; off >>= 1) amax = fmaxf(amax, __shfl_xor(amax, off, 32));
  if ((threadIdx.x & 31) == 0) wred[threadIdx.x >> 5] = amax;
  __syncthreads();
  if (threadIdx.x == 0) {
    float m = 0.f;
    for (int i = 0; i < 8; ++i) m = fmaxf(m, wred[i]);
    float s = fmaxf(m / 127.f, 1e-8f);
    sx[t] = s;
    sinv = 1.f / s;
  }
  __syncthreads();
  float inv = sinv;
  for (int c = threadIdx.x; c < CD; c += blockDim.x) {
    float q = rintf(row[c] * inv);
    q = fminf(fmaxf(q, -127.f), 127.f);
    xq[(size_t)t * CD + c] = (signed char)q;
  }
}

template <typename OutT>
__global__ void __launch_bounds__(256)
k_gemm_q8(const signed char* __restrict__ xq, const float* __restrict__ sx,
          const signed char* __restrict__ w, const float* __restrict__ ws,
          OutT* __restrict__ out) {
  __shared__ signed char lds_a[2][64][64];
  __shared__ signed char lds_b[2][64][64];
  const int lane = threadIdx.x & 31;
  const int wave = threadIdx.x >> 5;
  const int h16  = lane >> 4;
  const int ln   = lane & 15;
  const int tid  = threadIdx.x;

  int bm = blockIdx.x & 63;
  int bn = blockIdx.x >> 6;
  int m0 = bm * 64, n0 = bn * 64;

  const int srow = tid >> 2;
  const int scol = (tid & 3) * 16;
  const signed char* agp = xq + (size_t)(m0 + srow) * CD + scol;
  const signed char* bgp = w  + (size_t)(n0 + srow) * CD + scol;

  auto stage = [&](int kb, int buf) {
#if USE_ASYNC_LDS
    __builtin_amdgcn_global_load_async_to_lds_b128(
        (as1_v4i*)(signed char*)(agp + kb * 64),
        (as3_v4i*)&lds_a[buf][srow][scol], 0, 0);
    __builtin_amdgcn_global_load_async_to_lds_b128(
        (as1_v4i*)(signed char*)(bgp + kb * 64),
        (as3_v4i*)&lds_b[buf][srow][scol], 0, 0);
#else
    *(int4*)&lds_a[buf][srow][scol] = *(const int4*)(agp + kb * 64);
    *(int4*)&lds_b[buf][srow][scol] = *(const int4*)(bgp + kb * 64);
#endif
  };

  int akd[8], bkd[8];
#pragma unroll
  for (int v = 0; v < 8; ++v) {
    akd[v] = (v & 1) + ((v >> 1) & 1) * 4 + (v >> 2) * 8 + h16 * 2;
    bkd[v] = (v & 1) + ((v >> 1) & 1) * 4 + (v >> 2) * 8 + h16 * 2;
  }

  const int mtl = (wave & 3) * 16;
  const int ntl = (wave >> 2) * 32;

  v8i acc0 = {0, 0, 0, 0, 0, 0, 0, 0};
  v8i acc1 = {0, 0, 0, 0, 0, 0, 0, 0};

  stage(0, 0);
  for (int kb = 0; kb < CD / 64; ++kb) {
    int buf = kb & 1;
    if (kb + 1 < CD / 64) stage(kb + 1, buf ^ 1);
#if USE_ASYNC_LDS
    if (kb + 1 < CD / 64) wait_async_le2(); else wait_async_le0();
#endif
    __syncthreads();
    const unsigned* ar  = (const unsigned*)&lds_a[buf][mtl + ln][0];
    const unsigned* br0 = (const unsigned*)&lds_b[buf][ntl + ln][0];
    const unsigned* br1 = (const unsigned*)&lds_b[buf][ntl + 16 + ln][0];
    union { v8i v; unsigned u[8]; } a, b0, b1;
#pragma unroll
    for (int v = 0; v < 8; ++v) {
      a.u[v]  = ar[akd[v]];
      b0.u[v] = br0[bkd[v]];
      b1.u[v] = br1[bkd[v]];
    }
    acc0 = __builtin_amdgcn_wmma_i32_16x16x64_iu8(true, a.v, true, b0.v, acc0, false, false);
    acc1 = __builtin_amdgcn_wmma_i32_16x16x64_iu8(true, a.v, true, b1.v, acc1, false, false);
    __syncthreads();
  }

  float wsn0 = ws[n0 + ntl + ln];
  float wsn1 = ws[n0 + ntl + 16 + ln];
#pragma unroll
  for (int v = 0; v < 8; ++v) {
    int m = m0 + mtl + v + 8 * h16;
    float sm = sx[m];
    out[(size_t)m * CD + n0 + ntl + ln]      = (OutT)((float)acc0[v] * sm * wsn0);
    out[(size_t)m * CD + n0 + ntl + 16 + ln] = (OutT)((float)acc1[v] * sm * wsn1);
  }
}

__global__ void k_norm_rope(const _Float16* __restrict__ xin, const float* __restrict__ nw,
                            const float* __restrict__ fcos, const float* __restrict__ fsin,
                            _Float16* __restrict__ xout, int do_rope) {
  __shared__ float wred[8];
  __shared__ float sscale;
  int t = blockIdx.x;
  int b = t / CS, s = t - b * CS;
  const _Float16* row = xin + (size_t)t * CD;
  if (do_rope) {
    float ss = 0.f;
    for (int c = threadIdx.x; c < CD; c += blockDim.x) {
      float f = (float)row[c];
      ss += f * f;
    }
    for (int off = 16; off >= 1; off >>= 1) ss += __shfl_xor(ss, off, 32);
    if ((threadIdx.x & 31) == 0) wred[threadIdx.x >> 5] = ss;
    __syncthreads();
    if (threadIdx.x == 0) {
      float tot = 0.f;
      for (int i = 0; i < 8; ++i) tot += wred[i];
      sscale = rsqrtf(tot / (float)CD + 1e-6f);
    }
    __syncthreads();
  }
  float scale = do_rope ? sscale : 1.f;
  for (int p = threadIdx.x; p < CD / 2; p += blockDim.x) {
    int c0 = 2 * p;
    float x1 = (float)row[c0], x2 = (float)row[c0 + 1];
    int h0 = c0 >> 6, hd0 = c0 & 63;
    if (do_rope) {
      x1 *= scale * nw[c0];
      x2 *= scale * nw[c0 + 1];
      float cv = fcos[(size_t)s * (CD / 2) + p];
      float sv = fsin[(size_t)s * (CD / 2) + p];
      float o1 = x1 * cv - x2 * sv;
      float o2 = x1 * sv + x2 * cv;
      size_t base = (((size_t)b * CNH + h0) * CS + s) * CHD + hd0;
      xout[base]     = (_Float16)o1;
      xout[base + 1] = (_Float16)o2;
    } else {
      size_t base = (((size_t)b * CNH + h0) * CHD + hd0) * CS + s;
      xout[base]      = (_Float16)x1;
      xout[base + CS] = (_Float16)x2;
    }
  }
}

__global__ void __launch_bounds__(128)
k_attn(const _Float16* __restrict__ qh, const _Float16* __restrict__ kh,
       const _Float16* __restrict__ vt, const int* __restrict__ lens,
       float* __restrict__ out) {
  const int lane = threadIdx.x & 31;
  const int wv   = threadIdx.x >> 5;
  const int h16  = lane >> 4;
  const int ln   = lane & 15;
  int bid  = blockIdx.x;
  int qblk = bid & 31;
  int head = (bid >> 5) & 15;
  int b    = bid >> 9;
  int q0   = qblk * 64 + wv * 16;

  const _Float16* qbase = qh + (((size_t)b * CNH + head) * CS) * CHD;
  const _Float16* kbase = kh + (((size_t)b * CNH + head) * CS) * CHD;
  const _Float16* vbase = vt + (((size_t)b * CNH + head) * CHD) * CS;

  union F { v16h h; unsigned u[8]; _Float16 e[16]; };

  F qa0, qa1;
  {
    const unsigned* qrow = (const unsigned*)(qbase + (size_t)(q0 + ln) * CHD);
#pragma unroll
    for (int v = 0; v < 8; ++v) {
      int k = ((v < 4) ? 2 * v : 16 + 2 * (v - 4)) + h16 * 8;
      qa0.u[v] = qrow[k >> 1];
      qa1.u[v] = qrow[(k + 32) >> 1];
    }
  }

  float rmax[8], rsum[8];
  v8f o[4];
#pragma unroll
  for (int v = 0; v < 8; ++v) { rmax[v] = -1e30f; rsum[v] = 0.f; }
#pragma unroll
  for (int n = 0; n < 4; ++n) o[n] = (v8f){0, 0, 0, 0, 0, 0, 0, 0};

  __shared__ _Float16 pshare[4][16][32];

  int len = lens[b];
  int nkt = (len + 31) >> 5;
  for (int kt = 0; kt < nkt; ++kt) {
    int key0 = kt * 32;
    v8f s0, s1;
#pragma unroll
    for (int j = 0; j < 2; ++j) {
      F kb0, kb1;
      const unsigned* krow = (const unsigned*)(kbase + (size_t)(key0 + j * 16 + ln) * CHD);
#pragma unroll
      for (int v = 0; v < 8; ++v) {
        int k = ((v < 4) ? 2 * v : 16 + 2 * (v - 4)) + h16 * 8;
        kb0.u[v] = krow[k >> 1];
        kb1.u[v] = krow[(k + 32) >> 1];
      }
      v8f sc = {0, 0, 0, 0, 0, 0, 0, 0};
      sc = __builtin_amdgcn_wmma_f32_16x16x32_f16(false, qa0.h, false, kb0.h, (short)0, sc, false, false);
      sc = __builtin_amdgcn_wmma_f32_16x16x32_f16(false, qa1.h, false, kb1.h, (short)0, sc, false, false);
      if (j == 0) s0 = sc; else s1 = sc;
    }
    float mask0 = (key0 + ln      < len) ? 0.f : -1e30f;
    float mask1 = (key0 + 16 + ln < len) ? 0.f : -1e30f;
#pragma unroll
    for (int v = 0; v < 8; ++v) {
      s0[v] = s0[v] * 0.125f + mask0;
      s1[v] = s1[v] * 0.125f + mask1;
    }
#pragma unroll
    for (int v = 0; v < 8; ++v) {
      float t = row16_max(fmaxf(s0[v], s1[v]));
      float mnew = fmaxf(rmax[v], t);
      float corr = __expf(rmax[v] - mnew);
      rmax[v] = mnew;
      float p0 = __expf(s0[v] - mnew);
      float p1 = __expf(s1[v] - mnew);
      s0[v] = p0; s1[v] = p1;
      float ps = row16_sum(p0 + p1);
      rsum[v] = rsum[v] * corr + ps;
#pragma unroll
      for (int n = 0; n < 4; ++n) o[n][v] *= corr;
    }
#pragma unroll
    for (int v = 0; v < 8; ++v) {
      pshare[wv][v + 8 * h16][ln]      = (_Float16)s0[v];
      pshare[wv][v + 8 * h16][16 + ln] = (_Float16)s1[v];
    }
    F pa;
    {
      const unsigned* prow = (const unsigned*)&pshare[wv][ln][0];
#pragma unroll
      for (int v = 0; v < 8; ++v) {
        int kd = ((v < 4) ? v : v + 4) + h16 * 4;
        pa.u[v] = prow[kd];
      }
    }
#pragma unroll
    for (int n = 0; n < 4; ++n) {
      F vb;
      const unsigned* vrow = (const unsigned*)(vbase + (size_t)(n * 16 + ln) * CS + key0);
#pragma unroll
      for (int v = 0; v < 8; ++v) vb.u[v] = vrow[((v < 4) ? v : v + 4) + h16 * 4];
      o[n] = __builtin_amdgcn_wmma_f32_16x16x32_f16(false, pa.h, false, vb.h, (short)0, o[n], false, false);
    }
  }

#pragma unroll
  for (int v = 0; v < 8; ++v) {
    float inv = 1.f / rsum[v];
    int m = q0 + v + 8 * h16;
#pragma unroll
    for (int n = 0; n < 4; ++n) {
      out[((size_t)b * CS + m) * CD + head * CHD + n * 16 + ln] = o[n][v] * inv;
    }
  }
}

extern "C" void kernel_launch(void* const* d_in, const int* in_sizes, int n_in,
                              void* d_out, int out_size, void* d_ws, size_t ws_size,
                              hipStream_t stream) {
  (void)in_sizes; (void)n_in; (void)out_size; (void)ws_size;
  const float* hidden = (const float*)d_in[0];
  const float* fcos   = (const float*)d_in[1];
  const float* fsin   = (const float*)d_in[2];
  const int*   amask  = (const int*)d_in[3];
  const float* wq_q   = (const float*)d_in[4];
  const float* wq_s   = (const float*)d_in[5];
  const float* wk_q   = (const float*)d_in[6];
  const float* wk_s   = (const float*)d_in[7];
  const float* wv_q   = (const float*)d_in[8];
  const float* wv_s   = (const float*)d_in[9];
  const float* wo_q   = (const float*)d_in[10];
  const float* wo_s   = (const float*)d_in[11];
  const float* qnw    = (const float*)d_in[12];
  const float* knw    = (const float*)d_in[13];
  float* outp = (float*)d_out;

  char* ws = (char*)d_ws;
  size_t off = 0;
  auto alloc = [&](size_t bytes) -> void* {
    void* p = ws + off;
    off += (bytes + 255) & ~(size_t)255;
    return p;
  };
  const size_t WN = (size_t)CD * CD;
  signed char* w8   = (signed char*)alloc(4 * WN);
  signed char* xq   = (signed char*)alloc((size_t)CBS * CD);
  float*       sx   = (float*)alloc((size_t)CBS * 4);
  int*         lens = (int*)alloc(64);
  _Float16*    qlin = (_Float16*)alloc((size_t)CBS * CD * 2);
  _Float16*    klin = (_Float16*)alloc((size_t)CBS * CD * 2);
  _Float16*    vlin = (_Float16*)alloc((size_t)CBS * CD * 2);
  _Float16*    qhd  = (_Float16*)alloc((size_t)CBS * CD * 2);
  _Float16*    khd  = (_Float16*)alloc((size_t)CBS * CD * 2);
  _Float16*    vhdT = (_Float16*)alloc((size_t)CBS * CD * 2);
  float*       attn = (float*)alloc((size_t)CBS * CD * 4);
  signed char* aq   = (signed char*)alloc((size_t)CBS * CD);
  float*       asx  = (float*)alloc((size_t)CBS * 4);

  const int gemm_blocks = (CBS / 64) * (CD / 64);

  k_pack_w<<<(int)((WN + 255) / 256), 256, 0, stream>>>(wq_q, wk_q, wv_q, wo_q, w8);
  k_lengths<<<CB, 256, 0, stream>>>(amask, lens);
  k_quant<<<CBS, 256, 0, stream>>>(hidden, xq, sx);

  k_gemm_q8<_Float16><<<gemm_blocks, 256, 0, stream>>>(xq, sx, w8,          wq_s, qlin);
  k_gemm_q8<_Float16><<<gemm_blocks, 256, 0, stream>>>(xq, sx, w8 + WN,     wk_s, klin);
  k_gemm_q8<_Float16><<<gemm_blocks, 256, 0, stream>>>(xq, sx, w8 + 2 * WN, wv_s, vlin);

  k_norm_rope<<<CBS, 256, 0, stream>>>(qlin, qnw, fcos, fsin, qhd, 1);
  k_norm_rope<<<CBS, 256, 0, stream>>>(klin, knw, fcos, fsin, khd, 1);
  k_norm_rope<<<CBS, 256, 0, stream>>>(vlin, qnw, fcos, fsin, vhdT, 0);

  k_attn<<<CB * CNH * (CS / 64), 128, 0, stream>>>(qhd, khd, vhdT, lens, attn);

  k_quant<<<CBS, 256, 0, stream>>>(attn, aq, asx);
  k_gemm_q8<float><<<gemm_blocks, 256, 0, stream>>>(aq, asx, w8 + 3 * WN, wo_s, outp);
}
